// MoE_55551107007176
// MI455X (gfx1250) — matrix operand order repaired and checked by reading, not compiled, not run on hardware
//
#include <hip/hip_runtime.h>
#include <hip/hip_bf16.h>
#include <math.h>

typedef __attribute__((ext_vector_type(16))) _Float16 v16h;
typedef __attribute__((ext_vector_type(8)))  float    v8f;

namespace {
constexpr int kT  = 4096;   // tokens (2*2048)
constexpr int kH  = 1024;   // hidden
constexpr int kI  = 4096;   // intermediate
constexpr int kE  = 8;      // experts
constexpr int kMT = 64;     // token tile per block (4 WMMA m-subtiles)
constexpr int kKC = 128;    // I-chunk per iteration
}

union HFrag { v16h v; _Float16 h[16]; };
union FAcc  { v8f  v; float    f[8];  };

// ---------------------------------------------------------------- zero
__global__ void moe_zero_kernel(float* __restrict__ out, int n, int* __restrict__ cnt) {
  int i = blockIdx.x * blockDim.x + threadIdx.x;
  if (i < n) out[i] = 0.0f;
  if (blockIdx.x == 0 && threadIdx.x < kE) cnt[threadIdx.x] = 0;
}

// ---------------------------------------------------------------- router
// One wave (32 lanes) per token: 8 dot products of length 1024, lane-strided,
// cross-lane reduced with __shfl_xor (wave32). Lane 0 does top-2 + softmax,
// writes logits, and appends the token to its two expert lists.
__global__ void moe_router_kernel(const float* __restrict__ x,
                                  const float* __restrict__ gate_w,
                                  float* __restrict__ logits,
                                  int* __restrict__ cnt,
                                  int* __restrict__ lists,
                                  float* __restrict__ wts) {
  const int wave = threadIdx.x >> 5;
  const int lane = threadIdx.x & 31;
  const int t = blockIdx.x * (blockDim.x >> 5) + wave;
  if (t >= kT) return;

  const float* xr = x + (size_t)t * kH;
  float acc[kE];
  #pragma unroll
  for (int e = 0; e < kE; ++e) acc[e] = 0.0f;

  for (int h = lane; h < kH; h += 32) {
    float xv = xr[h];
    #pragma unroll
    for (int e = 0; e < kE; ++e) acc[e] += xv * gate_w[e * kH + h];
  }
  #pragma unroll
  for (int e = 0; e < kE; ++e) {
    float v = acc[e];
    #pragma unroll
    for (int off = 16; off > 0; off >>= 1) v += __shfl_xor(v, off, 32);
    acc[e] = v;
  }

  if (lane == 0) {
    #pragma unroll
    for (int e = 0; e < kE; ++e) logits[(size_t)t * kE + e] = acc[e];
    // top-1 (first occurrence on ties, matching jax.lax.top_k)
    int i0 = 0;
    for (int e = 1; e < kE; ++e) if (acc[e] > acc[i0]) i0 = e;
    // top-2
    int i1 = (i0 == 0) ? 1 : 0;
    for (int e = 0; e < kE; ++e) {
      if (e == i0) continue;
      if (acc[e] > acc[i1]) i1 = e;
    }
    float m  = fmaxf(acc[i0], acc[i1]);
    float e0 = __expf(acc[i0] - m);
    float e1 = __expf(acc[i1] - m);
    float inv = 1.0f / (e0 + e1);
    float w0 = e0 * inv, w1v = e1 * inv;

    int p0 = atomicAdd(&cnt[i0], 1);
    lists[i0 * kT + p0] = t;
    wts[i0 * kT + p0]   = w0;
    int p1 = atomicAdd(&cnt[i1], 1);
    lists[i1 * kT + p1] = t;
    wts[i1 * kT + p1]   = w1v;
  }
}

// helper: build a column-major B fragment from 16 contiguous fp32
__device__ __forceinline__ void load_b_frag(HFrag& b, const float* p) {
  float4 b0 = *(const float4*)(p);
  float4 b1 = *(const float4*)(p + 4);
  float4 b2 = *(const float4*)(p + 8);
  float4 b3 = *(const float4*)(p + 12);
  b.h[0]=(_Float16)b0.x;  b.h[1]=(_Float16)b0.y;  b.h[2]=(_Float16)b0.z;  b.h[3]=(_Float16)b0.w;
  b.h[4]=(_Float16)b1.x;  b.h[5]=(_Float16)b1.y;  b.h[6]=(_Float16)b1.z;  b.h[7]=(_Float16)b1.w;
  b.h[8]=(_Float16)b2.x;  b.h[9]=(_Float16)b2.y;  b.h[10]=(_Float16)b2.z; b.h[11]=(_Float16)b2.w;
  b.h[12]=(_Float16)b3.x; b.h[13]=(_Float16)b3.y; b.h[14]=(_Float16)b3.z; b.h[15]=(_Float16)b3.w;
}

// ---------------------------------------------------------------- fused FFN
// grid = (kT/kMT tiles, kE experts); block = 512 threads = 16 waves.
// Per block: gather 64 tokens routed to expert e, then for each 128-wide
// I-chunk: GEMM1 (x @ w1^T, K=1024) -> GELU -> LDS, GEMM2 accumulates the
// full 1024-wide output slice (16 waves x 64 columns).
// Four m-subtiles (rows 0-15,16-31,32-47,48-63) share every w2 B fragment.
__launch_bounds__(512)
__global__ void moe_ffn_kernel(const float* __restrict__ x,
                               const float* __restrict__ w1,
                               const float* __restrict__ w2,
                               const int* __restrict__ cnt,
                               const int* __restrict__ lists,
                               const float* __restrict__ wts,
                               float* __restrict__ out) {
  const int e = blockIdx.y;
  const int nrows = cnt[e];
  const int rowbase = blockIdx.x * kMT;
  if (rowbase >= nrows) return;

  __shared__ _Float16 xs[kMT][kH];   // 128 KB: gathered activations (f16)
  __shared__ _Float16 hs[kMT][kKC];  //  16 KB: GELU(h) chunk (f16)
  __shared__ int   stok[kMT];
  __shared__ float swt[kMT];

  const int tid = threadIdx.x;
  if (tid < kMT) {
    int idx = rowbase + tid;
    int tok = (idx < nrows) ? lists[e * kT + idx] : -1;
    stok[tid] = tok;
    swt[tid]  = (idx < nrows) ? wts[e * kT + idx] : 0.0f;
  }
  __syncthreads();

  // ---- cooperative gather of x tile (fp32 -> f16 in LDS) ----
  {
    int row = tid >> 3;           // 0..63
    int cg  = tid & 7;            // 8 column groups of 128
    int tok = stok[row];
    const float* xr = (tok >= 0) ? (x + (size_t)tok * kH) : nullptr;
    #pragma unroll
    for (int j = 0; j < 128; j += 4) {
      int col = cg * 128 + j;
      float4 v = (tok >= 0) ? *(const float4*)(xr + col) : make_float4(0.f, 0.f, 0.f, 0.f);
      xs[row][col + 0] = (_Float16)v.x;
      xs[row][col + 1] = (_Float16)v.y;
      xs[row][col + 2] = (_Float16)v.z;
      xs[row][col + 3] = (_Float16)v.w;
    }
  }
  __syncthreads();

  const int wave = tid >> 5;    // 0..15
  const int lane = tid & 31;
  const int hsel = lane >> 4;   // lane half (0/1)
  const int r    = lane & 15;

  v8f vzero = {};
  v8f oacc[4][4];
  #pragma unroll
  for (int mh = 0; mh < 4; ++mh)
    #pragma unroll
    for (int i = 0; i < 4; ++i) oacc[mh][i] = vzero;

  const float* w1e = w1 + (size_t)e * kI * kH;
  const float* w2e = w2 + (size_t)e * kH * kI;

  // GEMM1 work split: column-subtile cs = wave&7 (8 x 16 = 128 chunk cols),
  // m-pair mp = (wave>>3)*2 (waves 0-7 do rows 0-31, waves 8-15 do 32-63).
  const int cs = wave & 7;
  const int mp = (wave >> 3) * 2;
  // GEMM2 work split: wave owns output columns [wave*64, wave*64+64).
  const int n0w = wave * 64;

  for (int i0 = 0; i0 < kI; i0 += kKC) {
    // ---------------- GEMM1: two m-subtiles (mp, mp+1) for chunk columns
    // [i0 + cs*16, +16), K = kH = 1024; B fragment shared by both.
    v8f hacc0 = vzero, hacc1 = vzero;
    const float* w1row = w1e + (size_t)(i0 + cs * 16 + r) * kH;  // B column n=r
    for (int kk = 0; kk < kH; kk += 32) {
      HFrag a0, a1, b;
      // A fragments: lane row m=r; element e -> k = hsel*16 + e
      #pragma unroll
      for (int g = 0; g < 2; ++g) {
        int kb = kk + hsel * 16 + g * 8;
        #pragma unroll
        for (int j = 0; j < 8; ++j) {
          a0.h[g * 8 + j] = xs[mp * 16 + r][kb + j];
          a1.h[g * 8 + j] = xs[mp * 16 + 16 + r][kb + j];
        }
      }
      // B fragment (column-major): lane col n=r; element e -> k = hsel*16 + e
      load_b_frag(b, w1row + kk + hsel * 16);
      hacc0 = __builtin_amdgcn_wmma_f32_16x16x32_f16(
          false, a0.v, false, b.v, (short)0, hacc0, false, false);
      hacc1 = __builtin_amdgcn_wmma_f32_16x16x32_f16(
          false, a1.v, false, b.v, (short)0, hacc1, false, false);
    }

    // wait until every wave finished reading hs from the previous chunk
    __syncthreads();
    {
      FAcc hu0, hu1; hu0.v = hacc0; hu1.v = hacc1;
      #pragma unroll
      for (int v2 = 0; v2 < 8; ++v2) {
        int m = v2 + hsel * 8;                        // C/D row within subtile
        float h0 = hu0.f[v2];
        float h1 = hu1.f[v2];
        h0 = 0.5f * h0 * (1.0f + erff(h0 * 0.70710678118654752f));  // exact GELU
        h1 = 0.5f * h1 * (1.0f + erff(h1 * 0.70710678118654752f));
        hs[mp * 16 + m][cs * 16 + r]      = (_Float16)h0;  // C/D col n=r
        hs[mp * 16 + 16 + m][cs * 16 + r] = (_Float16)h1;
      }
    }
    __syncthreads();

    // ---------------- GEMM2: oacc[mh][t] += hs(64 x 128) @ w2 columns
    // A fragments hoisted per K-step; each B fragment feeds 4 m-subtiles.
    #pragma unroll
    for (int kc0 = 0; kc0 < kKC; kc0 += 32) {
      HFrag a[4];
      #pragma unroll
      for (int mh = 0; mh < 4; ++mh) {
        #pragma unroll
        for (int g = 0; g < 2; ++g) {
          int kb = kc0 + hsel * 16 + g * 8;
          #pragma unroll
          for (int j = 0; j < 8; ++j) a[mh].h[g * 8 + j] = hs[mh * 16 + r][kb + j];
        }
      }
      #pragma unroll
      for (int t2 = 0; t2 < 4; ++t2) {
        const float* w2row = w2e + (size_t)(n0w + t2 * 16 + r) * kI + i0;
        HFrag b;
        load_b_frag(b, w2row + kc0 + hsel * 16);
        #pragma unroll
        for (int mh = 0; mh < 4; ++mh) {
          oacc[mh][t2] = __builtin_amdgcn_wmma_f32_16x16x32_f16(
              false, a[mh].v, false, b.v, (short)0, oacc[mh][t2], false, false);
        }
      }
    }
  }

  // ---- apply router combine weight and accumulate into out ----
  #pragma unroll
  for (int mh = 0; mh < 4; ++mh) {
    #pragma unroll
    for (int t2 = 0; t2 < 4; ++t2) {
      FAcc ou; ou.v = oacc[mh][t2];
      #pragma unroll
      for (int v2 = 0; v2 < 8; ++v2) {
        int m = mh * 16 + v2 + hsel * 8;
        int tok = stok[m];
        if (tok >= 0) {
          float val = ou.f[v2] * swt[m];
          atomicAdd(out + (size_t)tok * kH + n0w + t2 * 16 + r, val);
        }
      }
    }
  }
}

// ---------------------------------------------------------------- launcher
extern "C" void kernel_launch(void* const* d_in, const int* in_sizes, int n_in,
                              void* d_out, int out_size, void* d_ws, size_t ws_size,
                              hipStream_t stream) {
  (void)in_sizes; (void)n_in; (void)out_size; (void)ws_size;
  const float* x      = (const float*)d_in[0];   // [2,2048,1024]
  const float* gate_w = (const float*)d_in[1];   // [8,1024]
  const float* w1     = (const float*)d_in[2];   // [8,4096,1024]
  const float* w2     = (const float*)d_in[3];   // [8,1024,4096]

  float* out    = (float*)d_out;                 // [T, H]
  float* logits = out + (size_t)kT * kH;         // [T, E]

  // workspace layout: cnt[16] | lists[E*T] (int) | wts[E*T] (float)  ~ 256 KB
  int*   cnt   = (int*)d_ws;
  int*   lists = cnt + 16;
  float* wts   = (float*)(lists + kE * kT);

  const int n = kT * kH;
  moe_zero_kernel<<<(n + 255) / 256, 256, 0, stream>>>(out, n, cnt);
  moe_router_kernel<<<kT / 8, 256, 0, stream>>>(x, gate_w, logits, cnt, lists, wts);
  dim3 grid(kT / kMT, kE);
  moe_ffn_kernel<<<grid, 512, 0, stream>>>(x, w1, w2, cnt, lists, wts, out);
}
